// NodeAttention_30124900614658
// MI455X (gfx1250) — matrix operand order repaired and checked by reading, not compiled, not run on hardware
//
#include <hip/hip_runtime.h>
#include <hip/hip_bf16.h>

typedef __attribute__((ext_vector_type(16))) _Float16 v16h;
typedef __attribute__((ext_vector_type(8)))  float    v8f;

#define CPN 32
#define HID 256
#define NTILES (HID / 16)        // 16 N-tiles of 16 cols
#define ROWS_PER_BLOCK 128       // 8 waves * 16 rows
#define WAVES_PER_BLOCK 8
#define LN_EPS 1e-5f

__global__ __launch_bounds__(256) void node_gate_fused(
    const float* __restrict__ x,
    const float* __restrict__ W1,
    const float* __restrict__ b1,
    const float* __restrict__ gamma,
    const float* __restrict__ beta,
    const float* __restrict__ W2,
    const float* __restrict__ b2p,
    float* __restrict__ out)
{
    // W1 staged as WMMA B-fragments (f16): 16 tiles * 32 lanes * 16 halfs
    __shared__ _Float16 ldsB[NTILES * 32 * 16];          // 16 KB
    __shared__ float ldsB1[HID], ldsG[HID], ldsBe[HID], ldsW2[HID]; // 4 KB
    __shared__ float ldsGate[WAVES_PER_BLOCK][16];

    const int tid     = threadIdx.x;
    const int lane    = tid & 31;
    const int wave    = tid >> 5;
    const int m       = lane & 15;     // row-in-tile (A) / col-in-tile (B,C,D)
    const int half_hi = lane >> 4;     // 0: lanes 0-15, 1: lanes 16-31

    // ---- Stage W1 into LDS as B fragments ----
    // B layout (32x16 K x N, 16-bit): lane n holds K {0..7, 16..23}, lane n+16 holds K {8..15, 24..31}.
    #pragma unroll
    for (int i = 0; i < (CPN * HID) / 256; ++i) {        // 32 iters
        int idx = tid + i * 256;                          // idx = k*256 + n (row-major [K][N])
        int k   = idx >> 8;
        int n   = idx & 255;
        int t   = n >> 4;
        int lt  = (n & 15) + ((k & 8) ? 16 : 0);
        int h   = (k & 7) + ((k & 16) ? 8 : 0);
        ldsB[(t * 32 + lt) * 16 + h] = (_Float16)W1[idx];
    }
    if (tid < HID) {
        ldsB1[tid] = b1[tid];
        ldsG[tid]  = gamma[tid];
        ldsBe[tid] = beta[tid];
        ldsW2[tid] = W2[tid];
    }
    __syncthreads();

    const int rowBase = blockIdx.x * ROWS_PER_BLOCK + wave * 16;

    // ---- Build A fragment (16 rows x 32 K, f16) ----
    // A layout (16-bit): lanes 0-15 (M=lane) hold K {0..7, 16..23};
    //                    lanes 16-31 (M=lane-16) hold K {8..15, 24..31}.
    const float* xrow = x + (size_t)(rowBase + m) * CPN;
    const int k0 = half_hi ? 8 : 0;
    float4 f0 = *(const float4*)(xrow + k0);
    float4 f1 = *(const float4*)(xrow + k0 + 4);
    float4 f2 = *(const float4*)(xrow + k0 + 16);
    float4 f3 = *(const float4*)(xrow + k0 + 20);
    v16h a;
    a[0]  = (_Float16)f0.x; a[1]  = (_Float16)f0.y; a[2]  = (_Float16)f0.z; a[3]  = (_Float16)f0.w;
    a[4]  = (_Float16)f1.x; a[5]  = (_Float16)f1.y; a[6]  = (_Float16)f1.z; a[7]  = (_Float16)f1.w;
    a[8]  = (_Float16)f2.x; a[9]  = (_Float16)f2.y; a[10] = (_Float16)f2.z; a[11] = (_Float16)f2.w;
    a[12] = (_Float16)f3.x; a[13] = (_Float16)f3.y; a[14] = (_Float16)f3.z; a[15] = (_Float16)f3.w;

    // ---- GEMM: h = x @ W1, one wmma per 16-col tile (K=32 in one op) ----
    v8f zero = {};
    v8f acc[NTILES];
    #pragma unroll
    for (int t = 0; t < NTILES; ++t) {
        v16h bf = *(const v16h*)&ldsB[(t * 32 + lane) * 16];
        acc[t] = __builtin_amdgcn_wmma_f32_16x16x32_f16(
            false, a, false, bf, (short)0, zero, false, false);
    }

    // ---- +b1, ELU, per-node sum / sumsq (C/D layout: VGPR v = row, lane%16 = col) ----
    float s[8], s2[8];
    #pragma unroll
    for (int v = 0; v < 8; ++v) { s[v] = 0.f; s2[v] = 0.f; }
    #pragma unroll
    for (int t = 0; t < NTILES; ++t) {
        int col  = t * 16 + m;
        float bb = ldsB1[col];
        #pragma unroll
        for (int v = 0; v < 8; ++v) {
            float hv = acc[t][v] + bb;
            hv = (hv > 0.f) ? hv : (__expf(hv) - 1.f);   // ELU
            acc[t][v] = hv;
            s[v]  += hv;
            s2[v] += hv * hv;
        }
    }
    // reduce across the 16-lane half (cols): each half owns its 8 rows
    #pragma unroll
    for (int v = 0; v < 8; ++v) {
        #pragma unroll
        for (int msk = 1; msk <= 8; msk <<= 1) {
            s[v]  += __shfl_xor(s[v],  msk, 32);
            s2[v] += __shfl_xor(s2[v], msk, 32);
        }
    }
    float mu[8], rs[8];
    #pragma unroll
    for (int v = 0; v < 8; ++v) {
        mu[v] = s[v] * (1.0f / HID);
        float var = s2[v] * (1.0f / HID) - mu[v] * mu[v];
        rs[v] = rsqrtf(var + LN_EPS);
    }

    // ---- LayerNorm affine + dot with W2 -> alpha per node ----
    float ap[8];
    #pragma unroll
    for (int v = 0; v < 8; ++v) ap[v] = 0.f;
    #pragma unroll
    for (int t = 0; t < NTILES; ++t) {
        int col  = t * 16 + m;
        float g  = ldsG[col];
        float be = ldsBe[col];
        float w  = ldsW2[col];
        #pragma unroll
        for (int v = 0; v < 8; ++v) {
            float hn = (acc[t][v] - mu[v]) * rs[v] * g + be;
            ap[v] += hn * w;
        }
    }
    #pragma unroll
    for (int v = 0; v < 8; ++v) {
        #pragma unroll
        for (int msk = 1; msk <= 8; msk <<= 1)
            ap[v] += __shfl_xor(ap[v], msk, 32);
    }

    const float b2v = b2p[0];
    float gate[8];
    #pragma unroll
    for (int v = 0; v < 8; ++v) {
        float alpha = ap[v] + b2v;
        gate[v] = 1.0f / (1.0f + __expf(-alpha));        // sigmoid
    }

    // broadcast the 16 per-node gates to all lanes of the wave via LDS
    if (m == 0) {                                         // lanes 0 and 16
        #pragma unroll
        for (int v = 0; v < 8; ++v)
            ldsGate[wave][half_hi * 8 + v] = gate[v];
    }
    __syncthreads();

    // ---- Gated output: out[row, c] = x[row, c] * gate[row] ----
    const float* xb = x   + (size_t)rowBase * CPN;
    float*       ob = out + (size_t)rowBase * CPN;
    #pragma unroll
    for (int j = 0; j < 16; ++j) {
        float g = ldsGate[wave][j];
        ob[j * CPN + lane] = xb[j * CPN + lane] * g;
    }
}

extern "C" void kernel_launch(void* const* d_in, const int* in_sizes, int n_in,
                              void* d_out, int out_size, void* d_ws, size_t ws_size,
                              hipStream_t stream) {
    const float* x     = (const float*)d_in[0];
    const float* W1    = (const float*)d_in[1];
    const float* b1    = (const float*)d_in[2];
    const float* gamma = (const float*)d_in[3];
    const float* beta  = (const float*)d_in[4];
    const float* W2    = (const float*)d_in[5];
    const float* b2    = (const float*)d_in[6];
    float* out = (float*)d_out;

    int rows = in_sizes[0] / CPN;                 // B * N_NODES = 262144
    dim3 grid(rows / ROWS_PER_BLOCK), block(256); // 2048 blocks of 8 waves
    node_gate_fused<<<grid, block, 0, stream>>>(x, W1, b1, gamma, beta, W2, b2, out);
}
